// RSVFiLM_39238821216430
// MI455X (gfx1250) — matrix operand order repaired and checked by reading, not compiled, not run on hardware
//
#include <hip/hip_runtime.h>

// ---- shapes (fixed by the reference) ----
#define B_    16
#define CF    256      // feature channels
#define HF    80
#define WF    80
#define CI    64       // illumination channels (D)
#define KEXP  3        // experts
#define HZ    40
#define WZ    40
#define NPIX  (HF*WF)  // 6400
#define KP    224      // GEMM K: 192 (K*D) + 3 (bias/P) + 29 zero pad -> multiple of 32
#define LDS_STRIDE 232 // padded LDS row stride (halves): 116 dwords -> conflict-free b128 reads

typedef __attribute__((ext_vector_type(16))) _Float16 v16h;
typedef __attribute__((ext_vector_type(8)))  _Float16 v8h;
typedef __attribute__((ext_vector_type(8)))  float    v8f;
typedef __attribute__((ext_vector_type(4)))  int      v4i;

typedef __attribute__((address_space(1))) v4i* gptr_v4i;  // global (__device__) int4*
typedef __attribute__((address_space(3))) v4i* lptr_v4i;  // LDS (__shared__) int4*

#if defined(__has_builtin)
#  if __has_builtin(__builtin_amdgcn_global_load_async_to_lds_b128)
#    define ASYNC_LDS 1
#  endif
#endif

// ---------------------------------------------------------------------------
// Kernel 1: pack expert weights + biases into Wmat[512][224] (f16).
// row r < 256  -> gamma channel r   ; row r >= 256 -> beta channel r-256
// col j < 192  -> W[k=j>>6][c][d=j&63] ; 192<=j<195 -> bias[k=j-192][c] ; else 0
// ---------------------------------------------------------------------------
__global__ void build_wmat(const float* __restrict__ Wg, const float* __restrict__ bg,
                           const float* __restrict__ Wb, const float* __restrict__ bb,
                           _Float16* __restrict__ Wmat) {
  int idx = blockIdx.x * blockDim.x + threadIdx.x;
  if (idx >= 512 * KP) return;
  int r = idx / KP, j = idx - r * KP;
  int c = r & 255;
  const float* W    = (r < 256) ? Wg : Wb;
  const float* bias = (r < 256) ? bg : bb;
  float v = 0.0f;
  if (j < 192) {
    int k = j >> 6, d = j & 63;
    v = W[(k * CF + c) * CI + d];
  } else if (j < 195) {
    v = bias[(j - 192) * CF + c];
  }
  Wmat[(size_t)r * KP + j] = (_Float16)v;
}

// ---------------------------------------------------------------------------
// Kernel 2: bilinear upsample (half-pixel centers) + build activation matrix
// Act[b][pix][j] (f16): j<192: P_k * Z_d ; 192..194: P_k ; 195..223: 0
// One block per (output row y, batch). Source rows y0/y1 staged in LDS.
// ---------------------------------------------------------------------------
__global__ void __launch_bounds__(256) build_act(const float* __restrict__ Z,
                                                 const float* __restrict__ P,
                                                 _Float16* __restrict__ Act) {
  const int y = blockIdx.x;   // 0..79
  const int b = blockIdx.y;   // 0..15

  __shared__ float zr[2][CI][WZ];    // 20 KB
  __shared__ float pr[2][KEXP][WZ];

  // vertical sample position: s = (y+0.5)*0.5 - 0.5
  float sy  = 0.5f * (float)y - 0.25f;
  float y0f = floorf(sy);
  float wy  = sy - y0f;
  int   yi  = (int)y0f;
  int   y0  = yi < 0 ? 0 : (yi > HZ - 1 ? HZ - 1 : yi);
  int   y1  = (yi + 1) < 0 ? 0 : ((yi + 1) > HZ - 1 ? HZ - 1 : yi + 1);

  for (int i = threadIdx.x; i < 2 * CI * WZ; i += blockDim.x) {
    int half = i / (CI * WZ);
    int rem  = i - half * (CI * WZ);
    int d = rem / WZ, x = rem - d * WZ;
    int yy = half ? y1 : y0;
    zr[half][d][x] = Z[(((size_t)b * CI + d) * HZ + yy) * WZ + x];
  }
  for (int i = threadIdx.x; i < 2 * KEXP * WZ; i += blockDim.x) {
    int half = i / (KEXP * WZ);
    int rem  = i - half * (KEXP * WZ);
    int k = rem / WZ, x = rem - k * WZ;
    int yy = half ? y1 : y0;
    pr[half][k][x] = P[(((size_t)b * KEXP + k) * HZ + yy) * WZ + x];
  }
  __syncthreads();

  for (int e = threadIdx.x; e < WF * KP; e += blockDim.x) {
    int w = e / KP, j = e - w * KP;
    float sx  = 0.5f * (float)w - 0.25f;
    float x0f = floorf(sx);
    float wx  = sx - x0f;
    int   xi  = (int)x0f;
    int   x0  = xi < 0 ? 0 : (xi > WZ - 1 ? WZ - 1 : xi);
    int   x1  = (xi + 1) < 0 ? 0 : ((xi + 1) > WZ - 1 ? WZ - 1 : xi + 1);

    float v = 0.0f;
    if (j < 195) {
      int k = (j < 192) ? (j >> 6) : (j - 192);
      float pup = (1.f - wy) * ((1.f - wx) * pr[0][k][x0] + wx * pr[0][k][x1])
                +        wy  * ((1.f - wx) * pr[1][k][x0] + wx * pr[1][k][x1]);
      if (j < 192) {
        int d = j & 63;
        float zup = (1.f - wy) * ((1.f - wx) * zr[0][d][x0] + wx * zr[0][d][x1])
                  +        wy  * ((1.f - wx) * zr[1][d][x0] + wx * zr[1][d][x1]);
        v = pup * zup;
      } else {
        v = pup;
      }
    }
    Act[((size_t)b * NPIX + (size_t)y * WF + w) * KP + j] = (_Float16)v;
  }
}

// ---------------------------------------------------------------------------
// Kernel 3: WMMA GEMM (M=512 gamma|beta rows, N=pixels, K=224) + fused FiLM.
// Block = 256 threads (8 waves). Block tile: 512 channels x 32 pixels.
// Activation tile (32 px x 224 K, 14 KB) is staged once per block in LDS
// via async global->LDS (ASYNCcnt) since all 8 waves consume identical
// B-fragments. Wave tile: 32 channels, computing BOTH gamma rows (c) and beta
// rows (c+256) so gamma/beta for the same (c,p) share lane/VGPR slots ->
// elementwise FiLM epilogue, no transpose, 64B-contiguous stores.
// ---------------------------------------------------------------------------

// A-operand (16x32 f16, M=rows of Wmat): per documented wave32 layout,
// lanes 0-15 hold K+{0..7} (elems 0-7) and K+16+{0..7} (elems 8-15);
// lanes 16-31 hold K+8+{0..7} and K+24+{0..7}. Row-major stride KP keeps
// both chunks 16B-aligned contiguous loads.
__device__ __forceinline__ v16h load_afrag(const _Float16* __restrict__ W,
                                           int row, int kb, int hi) {
  const _Float16* base = W + (size_t)row * KP + kb + hi * 8;
  v8h lo  = *(const v8h*)(base);
  v8h hi8 = *(const v8h*)(base + 16);
  v16h a;
#pragma unroll
  for (int i = 0; i < 8; ++i) { a[i] = lo[i]; a[i + 8] = hi8[i]; }
  return a;
}

// B-operand (32x16 f16, N=pixels) from the LDS tile: lane l16 = column,
// lane-half hi holds K+8*hi+{0..7} and K+16+8*hi+{0..7} -> two 16B ds_load chunks.
__device__ __forceinline__ v16h load_bfrag_lds(const _Float16* s, int row,
                                               int kb, int hi) {
  const _Float16* base = s + row * LDS_STRIDE + kb + hi * 8;
  v8h lo = *(const v8h*)(base);
  v8h h8 = *(const v8h*)(base + 16);
  v16h v;
#pragma unroll
  for (int i = 0; i < 8; ++i) { v[i] = lo[i]; v[i + 8] = h8[i]; }
  return v;
}

__global__ void __launch_bounds__(256) gemm_film(const _Float16* __restrict__ Act,
                                                 const _Float16* __restrict__ Wmat,
                                                 const float* __restrict__ feat,
                                                 float* __restrict__ out) {
  const int b    = blockIdx.y;
  const int p0   = blockIdx.x * 32;          // 2 pixel tiles of 16
  const int wave = threadIdx.x >> 5;         // 0..7
  const int lane = threadIdx.x & 31;
  const int l16  = lane & 15;
  const int hi   = lane >> 4;
  const int cbase = wave * 32;               // this wave's 32 channels

  __shared__ _Float16 sAct[32 * LDS_STRIDE]; // ~14.5 KB activation tile

  // ---- stage the 32x224 activation tile into LDS (once per block) ----
  {
    const _Float16* gbase = Act + ((size_t)b * NPIX + p0) * KP;
    // 32 rows x 28 chunks of 8 halves (16B) = 896 chunks
    for (int i = threadIdx.x; i < 32 * 28; i += 256) {
      int row = i / 28, ch = i - row * 28;
      const _Float16* gsrc = gbase + (size_t)row * KP + ch * 8;
      _Float16*       ldst = &sAct[row * LDS_STRIDE + ch * 8];
#ifdef ASYNC_LDS
      __builtin_amdgcn_global_load_async_to_lds_b128(
          (gptr_v4i)(void*)(_Float16*)gsrc, (lptr_v4i)(void*)ldst, 0, 0);
#else
      *(v8h*)ldst = *(const v8h*)gsrc;
#endif
    }
#ifdef ASYNC_LDS
#  if __has_builtin(__builtin_amdgcn_s_wait_asynccnt)
    __builtin_amdgcn_s_wait_asynccnt(0);
#  else
    asm volatile("s_wait_asynccnt 0x0" ::: "memory");
#  endif
#endif
    __syncthreads();
  }

  v8f accG[2][2];  // [channel-tile][pixel-tile] gamma
  v8f accB[2][2];  //                            beta
#pragma unroll
  for (int ct = 0; ct < 2; ++ct)
#pragma unroll
    for (int pt = 0; pt < 2; ++pt) {
      accG[ct][pt] = (v8f)(0.0f);
      accB[ct][pt] = (v8f)(0.0f);
    }

#pragma unroll
  for (int t = 0; t < KP / 32; ++t) {        // 7 K-steps of 32
    const int kb = t * 32;

    v16h bfrag[2];
#pragma unroll
    for (int pt = 0; pt < 2; ++pt)
      bfrag[pt] = load_bfrag_lds(sAct, pt * 16 + l16, kb, hi);

#pragma unroll
    for (int ct = 0; ct < 2; ++ct) {
      const int rowG = cbase + ct * 16 + l16;     // gamma row = channel
      v16h aG = load_afrag(Wmat, rowG,        kb, hi);
      v16h aB = load_afrag(Wmat, rowG + 256,  kb, hi);  // beta row
#pragma unroll
      for (int pt = 0; pt < 2; ++pt) {
        accG[ct][pt] = __builtin_amdgcn_wmma_f32_16x16x32_f16(
            false, aG, false, bfrag[pt], (short)0, accG[ct][pt], false, false);
        accB[ct][pt] = __builtin_amdgcn_wmma_f32_16x16x32_f16(
            false, aB, false, bfrag[pt], (short)0, accB[ct][pt], false, false);
      }
    }
  }

  // Fused FiLM epilogue: out = feat*(1+gamma)+beta.
  // C/D layout: elem r -> M = r + 8*hi (row/channel), N = l16 (pixel).
#pragma unroll
  for (int ct = 0; ct < 2; ++ct)
#pragma unroll
    for (int pt = 0; pt < 2; ++pt)
#pragma unroll
      for (int r = 0; r < 8; ++r) {
        int c = cbase + ct * 16 + r + hi * 8;
        int p = p0 + pt * 16 + l16;
        size_t idx = ((size_t)b * CF + c) * NPIX + p;
        float f = feat[idx];
        out[idx] = f * (1.0f + accG[ct][pt][r]) + accB[ct][pt][r];
      }
}

// ---------------------------------------------------------------------------
// Host launcher. Workspace: Wmat (229,376 B) then Act (45,875,200 B) ~= 46 MB.
// Inputs (setup_inputs order): feat, Z, P, Wg, bg, Wb, bb — all f32.
// ---------------------------------------------------------------------------
extern "C" void kernel_launch(void* const* d_in, const int* in_sizes, int n_in,
                              void* d_out, int out_size, void* d_ws, size_t ws_size,
                              hipStream_t stream) {
  const float* feat = (const float*)d_in[0];
  const float* Z    = (const float*)d_in[1];
  const float* P    = (const float*)d_in[2];
  const float* Wg   = (const float*)d_in[3];
  const float* bg   = (const float*)d_in[4];
  const float* Wb   = (const float*)d_in[5];
  const float* bb   = (const float*)d_in[6];
  float* out = (float*)d_out;

  _Float16* Wmat = (_Float16*)d_ws;
  _Float16* Act  = (_Float16*)((char*)d_ws + (size_t)512 * KP * sizeof(_Float16));

  build_wmat<<<(512 * KP + 255) / 256, 256, 0, stream>>>(Wg, bg, Wb, bb, Wmat);
  build_act<<<dim3(HF, B_), 256, 0, stream>>>(Z, P, Act);
  gemm_film<<<dim3(NPIX / 32, B_), 256, 0, stream>>>(Act, Wmat, feat, out);
}
